// HyperbolicMuRP_69303592288540
// MI455X (gfx1250) — compile-verified, matrix operand order checked by reading, not run on hardware
//
#include <hip/hip_runtime.h>
#include <hip/hip_bf16.h>

typedef _Float16 half8  __attribute__((ext_vector_type(8)));
typedef _Float16 half16 __attribute__((ext_vector_type(16)));
typedef float    float8 __attribute__((ext_vector_type(8)));

#define EPSF 1e-15f
#define MAXT (1.0f - 1e-5f)
#define CC   0.01f     // curvature c
#define SQC  0.1f      // sqrt(c)

static __device__ __forceinline__ float wave_sum(float v) {
    #pragma unroll
    for (int off = 16; off > 0; off >>= 1) v += __shfl_xor(v, off, 32);
    return v;
}

static __device__ __forceinline__ half16 combine16(half8 lo, half8 hi) {
    return __builtin_shufflevector(lo, hi, 0,1,2,3,4,5,6,7,8,9,10,11,12,13,14,15);
}

// ---------------------------------------------------------------------------
// Kernel 1: per-triplet query construction. One wave32 per triplet row,
// 4 fp32 elements per lane (D = 128). Writes query as f16 (WMMA A input) and
// qmeta[m] = { |query|^2 , entity_bias[s] } packed for one b64 epilogue load.
// ---------------------------------------------------------------------------
__global__ __launch_bounds__(256) void murp_query_kernel(
    const float* __restrict__ ent, const int* __restrict__ trip,
    const float* __restrict__ rdiag, const float* __restrict__ rtrans,
    const float* __restrict__ ebias,
    _Float16* __restrict__ qh, float2* __restrict__ qmeta,
    int B)
{
    const int row  = blockIdx.x * 8 + (threadIdx.x >> 5);
    const int lane = threadIdx.x & 31;
    if (row >= B) return;

    const int s = trip[row * 3 + 0];
    const int r = trip[row * 3 + 1];

    const float4 x  = ((const float4*)(ent    + (long)s * 128))[lane];
    const float4 rt = ((const float4*)(rdiag  + (long)r * 128))[lane];
    const float4 tr = ((const float4*)(rtrans + (long)r * 128))[lane];

    // log_map_zero(s_emb) * rot
    float n1 = fmaxf(sqrtf(wave_sum(x.x*x.x + x.y*x.y + x.z*x.z + x.w*x.w)), EPSF);
    float s1 = atanhf(fminf(SQC * n1, MAXT)) / (SQC * n1);
    float4 v; v.x = rt.x*(s1*x.x); v.y = rt.y*(s1*x.y); v.z = rt.z*(s1*x.z); v.w = rt.w*(s1*x.w);

    // exp_map_zero(v) -> rot_s
    float n2 = fmaxf(sqrtf(wave_sum(v.x*v.x + v.y*v.y + v.z*v.z + v.w*v.w)), EPSF);
    float s2 = tanhf(SQC * n2) / (SQC * n2);
    float4 xs; xs.x = s2*v.x; xs.y = s2*v.y; xs.z = s2*v.z; xs.w = s2*v.w;

    // exp_map_zero(rel_trans) -> t_r
    float n3 = fmaxf(sqrtf(wave_sum(tr.x*tr.x + tr.y*tr.y + tr.z*tr.z + tr.w*tr.w)), EPSF);
    float s3 = tanhf(SQC * n3) / (SQC * n3);
    float4 y; y.x = s3*tr.x; y.y = s3*tr.y; y.z = s3*tr.z; y.w = s3*tr.w;

    // mobius_add(rot_s, t_r, c)
    float x2 = wave_sum(xs.x*xs.x + xs.y*xs.y + xs.z*xs.z + xs.w*xs.w);
    float y2 = wave_sum(y.x*y.x + y.y*y.y + y.z*y.z + y.w*y.w);
    float xy = wave_sum(xs.x*y.x + xs.y*y.y + xs.z*y.z + xs.w*y.w);
    float ca  = 1.0f + 2.0f*CC*xy + CC*y2;
    float cb  = 1.0f - CC*x2;
    float den = fmaxf(1.0f + 2.0f*CC*xy + CC*CC*x2*y2, EPSF);
    float4 q; q.x = (ca*xs.x + cb*y.x)/den; q.y = (ca*xs.y + cb*y.y)/den;
              q.z = (ca*xs.z + cb*y.z)/den; q.w = (ca*xs.w + cb*y.w)/den;

    float q2 = wave_sum(q.x*q.x + q.y*q.y + q.z*q.z + q.w*q.w);

    _Float16* qp = qh + (long)row * 128 + lane * 4;
    qp[0] = (_Float16)q.x; qp[1] = (_Float16)q.y;
    qp[2] = (_Float16)q.z; qp[3] = (_Float16)q.w;
    if (lane == 0) { qmeta[row] = make_float2(q2, ebias[s]); }
}

// ---------------------------------------------------------------------------
// Kernel 2: WMMA GEMM (dot = query @ entityT) fused with hyperbolic-distance
// epilogue. One workgroup of 16 wave32s covers N-tile of 64 entities and ALL
// M = B rows, so each entity row is read from HBM exactly once.
//   wave wm = wid&3  -> M rows [wm*128, wm*128+128) as 8 subtiles of 16
//   wave wn = wid>>2 -> entity column subtile of 16
// Output stores use non-temporal hints: the 205 MB score matrix is streamed
// past L2 so the reused entity/query operands stay resident.
// ---------------------------------------------------------------------------
__global__ __launch_bounds__(512) void murp_gemm_kernel(
    const float* __restrict__ ent, const float* __restrict__ ebias,
    const _Float16* __restrict__ qh, const float2* __restrict__ qmeta,
    float* __restrict__ out,
    int E, int B)
{
    // padded stride 136 halves (=68 dwords) -> conflict-free ds_load_b128
    __shared__ __align__(16) _Float16 entH[64][136];
    __shared__ float partial[512];
    __shared__ float e2s[64];

    const int t  = threadIdx.x;
    const int N0 = blockIdx.x * 64;

    // ---- cooperative load: 64 entity rows fp32 -> f16 LDS, + e2 partials ----
    {
        const int n  = t >> 3;            // 0..63 row in tile
        const int c0 = (t & 7) * 16;      // 0..112 column start
        const long row = (long)min(N0 + n, E - 1);
        const float4* p = (const float4*)(ent + row * 128 + c0);
        float4 f0 = p[0], f1 = p[1], f2 = p[2], f3 = p[3];
        float ss = f0.x*f0.x + f0.y*f0.y + f0.z*f0.z + f0.w*f0.w
                 + f1.x*f1.x + f1.y*f1.y + f1.z*f1.z + f1.w*f1.w
                 + f2.x*f2.x + f2.y*f2.y + f2.z*f2.z + f2.w*f2.w
                 + f3.x*f3.x + f3.y*f3.y + f3.z*f3.z + f3.w*f3.w;
        partial[t] = ss;
        half8 h0 = {(_Float16)f0.x,(_Float16)f0.y,(_Float16)f0.z,(_Float16)f0.w,
                    (_Float16)f1.x,(_Float16)f1.y,(_Float16)f1.z,(_Float16)f1.w};
        half8 h1 = {(_Float16)f2.x,(_Float16)f2.y,(_Float16)f2.z,(_Float16)f2.w,
                    (_Float16)f3.x,(_Float16)f3.y,(_Float16)f3.z,(_Float16)f3.w};
        *(half8*)&entH[n][c0]     = h0;
        *(half8*)&entH[n][c0 + 8] = h1;
    }
    __syncthreads();
    if (t < 64) {
        float s = 0.f;
        #pragma unroll
        for (int i = 0; i < 8; ++i) s += partial[t * 8 + i];
        e2s[t] = s;
    }
    __syncthreads();

    const int wid = t >> 5, l = t & 31;
    const int wm = wid & 3, wn = wid >> 2;
    const int nl = l & 15;                 // row-within-tile / column index
    const int kh = (l >> 4) * 8;           // K half-chunk select per §7.12.2

    // ---- B operands: 4 K-steps of entity columns from LDS ----
    half16 bops[4];
    #pragma unroll
    for (int ks = 0; ks < 4; ++ks) {
        const _Float16* bp = &entH[wn * 16 + nl][ks * 32 + kh];
        bops[ks] = combine16(*(const half8*)bp, *(const half8*)(bp + 16));
    }

    // ---- 8 M-subtiles x 4 K-steps of WMMA ----
    float8 acc[8] = {};
    #pragma unroll
    for (int mt = 0; mt < 8; ++mt) {
        const int mrow = min(wm * 128 + mt * 16 + nl, B - 1);
        const _Float16* ap = qh + (long)mrow * 128 + kh;
        #pragma unroll
        for (int ks = 0; ks < 4; ++ks) {
            half16 a = combine16(*(const half8*)(ap + ks * 32),
                                 *(const half8*)(ap + ks * 32 + 16));
            acc[mt] = __builtin_amdgcn_wmma_f32_16x16x32_f16(
                false, a, false, bops[ks], (short)0, acc[mt], false, false);
        }
    }

    // ---- epilogue: Poincare squared distance + biases ----
    const int n   = N0 + wn * 16 + nl;
    const float e2n = e2s[wn * 16 + nl];
    const float ebn = ebias[min(n, E - 1)];
    const int moff = (l >> 4) << 3;        // lanes 16-31 hold M+8 (§7.12.2 C layout)

    #pragma unroll
    for (int mt = 0; mt < 8; ++mt) {
        const int mb = wm * 128 + mt * 16;
        #pragma unroll
        for (int r = 0; r < 8; ++r) {
            const int m  = mb + r + moff;
            const int mc = min(m, B - 1);
            const float dot = acc[mt][r];
            const float2 qm = qmeta[mc];          // {q2, sbias} one b64 load
            const float q2m = qm.x;
            const float Af  = 1.0f - 2.0f*CC*dot + CC*e2n;
            const float Bq  = 1.0f - CC*q2m;
            const float dn  = fmaxf(1.0f - 2.0f*CC*dot + CC*CC*q2m*e2n, EPSF);
            const float nm2 = Af*Af*q2m + Bq*Bq*e2n - 2.0f*Af*Bq*dot;
            const float sc  = -(nm2 / (dn * dn)) + ebn + qm.y;
            if (m < B && n < E)
                __builtin_nontemporal_store(sc, &out[(size_t)m * (size_t)E + n]);
        }
    }
}

// ---------------------------------------------------------------------------
extern "C" void kernel_launch(void* const* d_in, const int* in_sizes, int n_in,
                              void* d_out, int out_size, void* d_ws, size_t ws_size,
                              hipStream_t stream) {
    const float* ent    = (const float*)d_in[0];
    // d_in[1] = rel_embedding (unused by the reference)
    const int*   trip   = (const int*)d_in[2];
    const float* rdiag  = (const float*)d_in[3];
    const float* rtrans = (const float*)d_in[4];
    const float* ebias  = (const float*)d_in[5];

    const int E = in_sizes[0] / 128;   // 100000
    const int B = in_sizes[2] / 3;     // 512
    float* out  = (float*)d_out;

    // workspace layout: f16 query [B][128] | qmeta float2[B]
    char* ws = (char*)d_ws;
    _Float16* qh = (_Float16*)ws;
    size_t qh_bytes = (((size_t)B * 128 * sizeof(_Float16)) + 255) & ~(size_t)255;
    float2* qmeta = (float2*)(ws + qh_bytes);

    murp_query_kernel<<<(B + 7) / 8, 256, 0, stream>>>(
        ent, trip, rdiag, rtrans, ebias, qh, qmeta, B);

    murp_gemm_kernel<<<(E + 63) / 64, 512, 0, stream>>>(
        ent, ebias, qh, qmeta, out, E, B);
}
